// Attention_54460185313727
// MI455X (gfx1250) — compile-verified, matrix operand order checked by reading, not run on hardware
//
#include <hip/hip_runtime.h>
#include <math.h>

typedef __attribute__((ext_vector_type(16))) _Float16 v16h;
typedef __attribute__((ext_vector_type(2)))  _Float16 h2;
typedef __attribute__((ext_vector_type(8)))  float    v8f;
typedef int v4i __attribute__((vector_size(16)));

#define D_DIM 64
#define S_LEN 2048
#define B_N   16
#define QT    16
#define KT    64
#define WAVES 4
#define NKB   (S_LEN / KT)          // 32 key tiles
#define NQT   (S_LEN / QT)          // 128 query tiles
#define TENH  (B_N * S_LEN * D_DIM) // halves per tensor in ws = 2^21
#define WS_BYTES (3u * TENH * 2u)   // 12 MB

// 16-bit A/B operand swizzle (CDNA5 ISA §7.12.2):
// lane = row + 16*half, element e;  k = {e[3]->bit4, half->bit3, e[2:0]}
__device__ __forceinline__ int lmapE(int k)    { return (k & 7) | ((k >> 1) & 8); }
__device__ __forceinline__ int lmapHalf(int k) { return (k >> 3) & 1; }
// inverse: given even element e and lane-half -> low 5 bits of k
__device__ __forceinline__ int kinv(int e, int half) {
  return (e & 7) | (half << 3) | ((e & 8) << 1);
}

__device__ __forceinline__ v8f wmma16(v16h a, v16h b, v8f c) {
  return __builtin_amdgcn_wmma_f32_16x16x32_f16(false, a, false, b, (short)0, c,
                                                false, false);
}

#define SWIZ_XOR(x, m) \
  __int_as_float(__builtin_amdgcn_ds_swizzle(__float_as_int(x), (0x1f | ((m) << 10))))

// ---- 16-byte global->LDS copy: async (ASYNCcnt) if toolchain has it ----
#if __has_builtin(__builtin_amdgcn_global_load_async_to_lds_b128)
#define HAS_ASYNC_LDS 1
__device__ __forceinline__ void copy16(const _Float16* g, _Float16* l) {
  __builtin_amdgcn_global_load_async_to_lds_b128(
      (__attribute__((address_space(1))) v4i*)g,
      (__attribute__((address_space(3))) v4i*)l, 0, 0);
}
#else
#define HAS_ASYNC_LDS 0
__device__ __forceinline__ void copy16(const _Float16* g, _Float16* l) {
  *(uint4*)l = *(const uint4*)g;   // global_load_b128 + ds_store_b128
}
#endif
__device__ __forceinline__ void wait_async() {
  asm volatile("s_wait_asynccnt 0x0" ::: "memory");
}

// =====================================================================
// Pre-pass: fp32 [D,S,B] -> fp16 ws, already in WMMA operand tile order.
//  qws: [b][qt(128)][dH(2)][lane(32)][e(16)]       (Q pre-scaled)
//  kws: [b][kb(32)][set=grp*2+dH (8)][lane][e]     (B operand of Q*K^T)
//  vws: [b][kb(32)][set=kh*4+c  (8)][lane][e]      (B operand of P*V)
// Each thread emits one aligned 4-byte pair (e even, e+1).
// =====================================================================
__global__ __launch_bounds__(256)
void cvt_kernel(const float* __restrict__ Q, const float* __restrict__ K,
                const float* __restrict__ V, _Float16* __restrict__ ws,
                float scale) {
  unsigned t = blockIdx.x * 256u + threadIdx.x;   // 0 .. 3*2^20-1 (pairs)
  unsigned tensor = t >> 20;
  unsigned f = (t & ((1u << 20) - 1)) << 1;       // even half-index
  int e    = f & 15;
  int lane = (f >> 4) & 31;
  unsigned rest = f >> 9;
  int half = lane >> 4, low = lane & 15;
  int kk0  = kinv(e, half);                       // low-5 k for element e

  float v0, v1;
  if (tensor == 0) {          // Q
    int dH = rest & 1; int qt = (rest >> 1) & 127; int b = rest >> 8;
    int d = dH * 32 + kk0;    int s = qt * QT + low;
    v0 = Q[((size_t)d * S_LEN + s) * B_N + b] * scale;
    v1 = Q[((size_t)(d + 1) * S_LEN + s) * B_N + b] * scale;
  } else if (tensor == 1) {   // K
    int set = rest & 7; int kb = (rest >> 3) & 31; int b = rest >> 8;
    int grp = set >> 1, dH = set & 1;
    int d = dH * 32 + kk0;    int s = kb * KT + grp * 16 + low;
    v0 = K[((size_t)d * S_LEN + s) * B_N + b];
    v1 = K[((size_t)(d + 1) * S_LEN + s) * B_N + b];
  } else {                    // V
    int set = rest & 7; int kb = (rest >> 3) & 31; int b = rest >> 8;
    int kh = set >> 2, c = set & 3;
    int d = c * 16 + low;     int s = kb * KT + kh * 32 + kk0;
    v0 = V[((size_t)d * S_LEN + s) * B_N + b];
    v1 = V[((size_t)d * S_LEN + (s + 1)) * B_N + b];
  }
  h2 out; out[0] = (_Float16)v0; out[1] = (_Float16)v1;
  *(h2*)(ws + (size_t)tensor * TENH + f) = out;
}

// =====================================================================
// Main flash-attention kernel: async double-buffered LDS staging + WMMA
// =====================================================================
__global__ __launch_bounds__(128)
void fa_ws_kernel(const _Float16* __restrict__ ws, float* __restrict__ Out) {
  __shared__ alignas(32) _Float16 qbuf[4096];       // 8 KB
  __shared__ alignas(32) _Float16 kbuf[2][4096];    // 16 KB, double buffer
  __shared__ alignas(32) _Float16 vbuf[2][4096];    // 16 KB, double buffer
  __shared__ alignas(32) _Float16 pbuf[4096];       // 8 KB

  const _Float16* qws = ws;
  const _Float16* kws = ws + TENH;
  const _Float16* vws = ws + 2 * (size_t)TENH;

  const int tid    = threadIdx.x;
  const int wave   = tid >> 5;
  const int lane   = tid & 31;
  const int bid    = blockIdx.x;
  const int b      = bid >> 5;
  const int qtile0 = (bid & 31) * WAVES;
  const int qBase  = (qtile0 + wave) * QT;

  // ---- issue Q tiles + key tile 0 copies (pure data movement) ----
  {
    const _Float16* qsrc = qws + ((size_t)(b * NQT + qtile0)) * 1024;
    const _Float16* ksrc = kws + ((size_t)(b * NKB + 0)) * 4096;
    const _Float16* vsrc = vws + ((size_t)(b * NKB + 0)) * 4096;
#pragma unroll
    for (int i = 0; i < 4; ++i) {
      int off = (tid + 128 * i) * 8;        // 16B chunks
      copy16(qsrc + off, qbuf + off);
      copy16(ksrc + off, kbuf[0] + off);
      copy16(vsrc + off, vbuf[0] + off);
    }
  }
  wait_async();
  __syncthreads();

  const v16h* qv16 = (const v16h*)qbuf;
  const v16h* pv16 = (const v16h*)pbuf;
  const v16h qa0 = qv16[(wave * 2 + 0) * 32 + lane];   // D 0..31
  const v16h qa1 = qv16[(wave * 2 + 1) * 32 + lane];   // D 32..63

  v16h vones;
#pragma unroll
  for (int e = 0; e < 16; ++e) vones[e] = (_Float16)1.0f;

  v8f o0 = {}, o1 = {}, o2 = {}, o3 = {};
  float mrun[8], lrun[8];
#pragma unroll
  for (int r = 0; r < 8; ++r) { mrun[r] = -3.0e38f; lrun[r] = 0.0f; }

  const int halfId = lane >> 4;
  const int nCol   = lane & 15;

  for (int kb = 0; kb < NKB; ++kb) {
    const int cur = kb & 1, nxt = cur ^ 1;
    wait_async();        // tile kb resident in buffer `cur`
    __syncthreads();     // all waves done copying kb / done reading buffer nxt

    if (kb + 1 < NKB) {  // stream tile kb+1 while computing tile kb
      const _Float16* ksrc = kws + ((size_t)(b * NKB + kb + 1)) * 4096;
      const _Float16* vsrc = vws + ((size_t)(b * NKB + kb + 1)) * 4096;
#pragma unroll
      for (int i = 0; i < 4; ++i) {
        int off = (tid + 128 * i) * 8;
        copy16(ksrc + off, kbuf[nxt] + off);
        copy16(vsrc + off, vbuf[nxt] + off);
      }
    }

    const v16h* kv16 = (const v16h*)kbuf[cur];
    const v16h* vv16 = (const v16h*)vbuf[cur];

    // ---- S = Q K^T : four 16x16 score tiles ----
    v8f cs0 = {}, cs1 = {}, cs2 = {}, cs3 = {};
    cs0 = wmma16(qa0, kv16[0 * 32 + lane], cs0);
    cs0 = wmma16(qa1, kv16[1 * 32 + lane], cs0);
    cs1 = wmma16(qa0, kv16[2 * 32 + lane], cs1);
    cs1 = wmma16(qa1, kv16[3 * 32 + lane], cs1);
    cs2 = wmma16(qa0, kv16[4 * 32 + lane], cs2);
    cs2 = wmma16(qa1, kv16[5 * 32 + lane], cs2);
    cs3 = wmma16(qa0, kv16[6 * 32 + lane], cs3);
    cs3 = wmma16(qa1, kv16[7 * 32 + lane], cs3);

    // ---- online softmax: row max via ds_swizzle; P -> LDS (A layout) ----
    float al[8];
#pragma unroll
    for (int r = 0; r < 8; ++r) {
      float rm = fmaxf(fmaxf(cs0[r], cs1[r]), fmaxf(cs2[r], cs3[r]));
      rm = fmaxf(rm, SWIZ_XOR(rm, 1));
      rm = fmaxf(rm, SWIZ_XOR(rm, 2));
      rm = fmaxf(rm, SWIZ_XOR(rm, 4));
      rm = fmaxf(rm, SWIZ_XOR(rm, 8));
      float mnew = fmaxf(mrun[r], rm);
      al[r] = __expf(mrun[r] - mnew);
      mrun[r] = mnew;
      float p0 = __expf(cs0[r] - mnew);
      float p1 = __expf(cs1[r] - mnew);
      float p2 = __expf(cs2[r] - mnew);
      float p3 = __expf(cs3[r] - mnew);
      o0[r] *= al[r]; o1[r] *= al[r]; o2[r] *= al[r]; o3[r] *= al[r];

      int m = r + 8 * halfId;
      int kkA = nCol, kkB = nCol + 16;
      pbuf[(((wave * 2 + 0) * 32) + (m + 16 * lmapHalf(kkA))) * 16 + lmapE(kkA)] = (_Float16)p0;
      pbuf[(((wave * 2 + 0) * 32) + (m + 16 * lmapHalf(kkB))) * 16 + lmapE(kkB)] = (_Float16)p1;
      pbuf[(((wave * 2 + 1) * 32) + (m + 16 * lmapHalf(kkA))) * 16 + lmapE(kkA)] = (_Float16)p2;
      pbuf[(((wave * 2 + 1) * 32) + (m + 16 * lmapHalf(kkB))) * 16 + lmapE(kkB)] = (_Float16)p3;
    }
    asm volatile("s_wait_dscnt 0" ::: "memory");  // wave-local P RAW hazard

    v16h pa0 = pv16[(wave * 2 + 0) * 32 + lane];
    v16h pa1 = pv16[(wave * 2 + 1) * 32 + lane];

    // ---- row sums via WMMA against all-ones B ----
    v8f csum = {};
    csum = wmma16(pa0, vones, csum);
    csum = wmma16(pa1, vones, csum);
#pragma unroll
    for (int r = 0; r < 8; ++r) lrun[r] = lrun[r] * al[r] + csum[r];

    // ---- O += P V ----
    o0 = wmma16(pa0, vv16[0 * 32 + lane], o0);
    o1 = wmma16(pa0, vv16[1 * 32 + lane], o1);
    o2 = wmma16(pa0, vv16[2 * 32 + lane], o2);
    o3 = wmma16(pa0, vv16[3 * 32 + lane], o3);
    o0 = wmma16(pa1, vv16[4 * 32 + lane], o0);
    o1 = wmma16(pa1, vv16[5 * 32 + lane], o1);
    o2 = wmma16(pa1, vv16[6 * 32 + lane], o2);
    o3 = wmma16(pa1, vv16[7 * 32 + lane], o3);
  }

  // ---- normalize and scatter back to [D, S, B] ----
#pragma unroll
  for (int r = 0; r < 8; ++r) {
    float inv = 1.0f / lrun[r];
    int s = qBase + r + 8 * halfId;
    Out[((size_t)(nCol +  0) * S_LEN + s) * B_N + b] = o0[r] * inv;
    Out[((size_t)(nCol + 16) * S_LEN + s) * B_N + b] = o1[r] * inv;
    Out[((size_t)(nCol + 32) * S_LEN + s) * B_N + b] = o2[r] * inv;
    Out[((size_t)(nCol + 48) * S_LEN + s) * B_N + b] = o3[r] * inv;
  }
}

// =====================================================================
// Fallback (workspace too small): self-contained kernel, converts in-loop
// =====================================================================
__global__ __launch_bounds__(128)
void fa_fallback_kernel(const float* __restrict__ Q, const float* __restrict__ K,
                        const float* __restrict__ V, float* __restrict__ Out,
                        float scale) {
  __shared__ alignas(32) _Float16 qbuf[8 * 32 * 16];
  __shared__ alignas(32) _Float16 kbuf[8 * 32 * 16];
  __shared__ alignas(32) _Float16 vbuf[8 * 32 * 16];
  __shared__ alignas(32) _Float16 pbuf[8 * 32 * 16];

  const int tid = threadIdx.x, wave = tid >> 5, lane = tid & 31;
  const int bid = blockIdx.x, b = bid >> 5;
  const int qtile0 = (bid & 31) * WAVES;
  const int qBase0 = qtile0 * QT, qBase = qBase0 + wave * QT;

  for (int i = 0; i < 32; ++i) {
    int idx = tid + 128 * i;
    int d = idx & 63, lq = idx >> 6, w = lq >> 4, m = lq & 15;
    float qv = Q[((size_t)d * S_LEN + (qBase0 + lq)) * B_N + b] * scale;
    int dH = d >> 5, dd = d & 31;
    qbuf[(((w * 2 + dH) * 32) + (m + 16 * lmapHalf(dd))) * 16 + lmapE(dd)] = (_Float16)qv;
  }
  __syncthreads();

  const v16h* qv16 = (const v16h*)qbuf;
  const v16h* kv16 = (const v16h*)kbuf;
  const v16h* vv16 = (const v16h*)vbuf;
  const v16h* pv16 = (const v16h*)pbuf;
  const v16h qa0 = qv16[(wave * 2 + 0) * 32 + lane];
  const v16h qa1 = qv16[(wave * 2 + 1) * 32 + lane];
  v16h vones;
#pragma unroll
  for (int e = 0; e < 16; ++e) vones[e] = (_Float16)1.0f;

  v8f o0 = {}, o1 = {}, o2 = {}, o3 = {};
  float mrun[8], lrun[8];
#pragma unroll
  for (int r = 0; r < 8; ++r) { mrun[r] = -3.0e38f; lrun[r] = 0.0f; }
  const int halfId = lane >> 4, nCol = lane & 15;

  for (int kb = 0; kb < NKB; ++kb) {
    const int s0 = kb * KT;
    __syncthreads();
    for (int i = 0; i < 32; ++i) {
      int idx = tid + 128 * i;
      int j = idx & 63, d = idx >> 6;
      size_t g = ((size_t)d * S_LEN + (s0 + j)) * B_N + b;
      int grp = j >> 4, n = j & 15, dH = d >> 5, dd = d & 31;
      kbuf[(((grp * 2 + dH) * 32) + (n + 16 * lmapHalf(dd))) * 16 + lmapE(dd)] = (_Float16)K[g];
      int kh = j >> 5, jj = j & 31, c = d >> 4, nv = d & 15;
      vbuf[(((kh * 4 + c) * 32) + (nv + 16 * lmapHalf(jj))) * 16 + lmapE(jj)] = (_Float16)V[g];
    }
    __syncthreads();

    v8f cs0 = {}, cs1 = {}, cs2 = {}, cs3 = {};
    cs0 = wmma16(qa0, kv16[0 * 32 + lane], cs0);
    cs0 = wmma16(qa1, kv16[1 * 32 + lane], cs0);
    cs1 = wmma16(qa0, kv16[2 * 32 + lane], cs1);
    cs1 = wmma16(qa1, kv16[3 * 32 + lane], cs1);
    cs2 = wmma16(qa0, kv16[4 * 32 + lane], cs2);
    cs2 = wmma16(qa1, kv16[5 * 32 + lane], cs2);
    cs3 = wmma16(qa0, kv16[6 * 32 + lane], cs3);
    cs3 = wmma16(qa1, kv16[7 * 32 + lane], cs3);

    float al[8];
#pragma unroll
    for (int r = 0; r < 8; ++r) {
      float rm = fmaxf(fmaxf(cs0[r], cs1[r]), fmaxf(cs2[r], cs3[r]));
      rm = fmaxf(rm, SWIZ_XOR(rm, 1));
      rm = fmaxf(rm, SWIZ_XOR(rm, 2));
      rm = fmaxf(rm, SWIZ_XOR(rm, 4));
      rm = fmaxf(rm, SWIZ_XOR(rm, 8));
      float mnew = fmaxf(mrun[r], rm);
      al[r] = __expf(mrun[r] - mnew);
      mrun[r] = mnew;
      float p0 = __expf(cs0[r] - mnew);
      float p1 = __expf(cs1[r] - mnew);
      float p2 = __expf(cs2[r] - mnew);
      float p3 = __expf(cs3[r] - mnew);
      o0[r] *= al[r]; o1[r] *= al[r]; o2[r] *= al[r]; o3[r] *= al[r];
      int m = r + 8 * halfId;
      int kkA = nCol, kkB = nCol + 16;
      pbuf[(((wave * 2 + 0) * 32) + (m + 16 * lmapHalf(kkA))) * 16 + lmapE(kkA)] = (_Float16)p0;
      pbuf[(((wave * 2 + 0) * 32) + (m + 16 * lmapHalf(kkB))) * 16 + lmapE(kkB)] = (_Float16)p1;
      pbuf[(((wave * 2 + 1) * 32) + (m + 16 * lmapHalf(kkA))) * 16 + lmapE(kkA)] = (_Float16)p2;
      pbuf[(((wave * 2 + 1) * 32) + (m + 16 * lmapHalf(kkB))) * 16 + lmapE(kkB)] = (_Float16)p3;
    }
    asm volatile("s_wait_dscnt 0" ::: "memory");

    v16h pa0 = pv16[(wave * 2 + 0) * 32 + lane];
    v16h pa1 = pv16[(wave * 2 + 1) * 32 + lane];
    v8f csum = {};
    csum = wmma16(pa0, vones, csum);
    csum = wmma16(pa1, vones, csum);
#pragma unroll
    for (int r = 0; r < 8; ++r) lrun[r] = lrun[r] * al[r] + csum[r];

    o0 = wmma16(pa0, vv16[0 * 32 + lane], o0);
    o1 = wmma16(pa0, vv16[1 * 32 + lane], o1);
    o2 = wmma16(pa0, vv16[2 * 32 + lane], o2);
    o3 = wmma16(pa0, vv16[3 * 32 + lane], o3);
    o0 = wmma16(pa1, vv16[4 * 32 + lane], o0);
    o1 = wmma16(pa1, vv16[5 * 32 + lane], o1);
    o2 = wmma16(pa1, vv16[6 * 32 + lane], o2);
    o3 = wmma16(pa1, vv16[7 * 32 + lane], o3);
  }
#pragma unroll
  for (int r = 0; r < 8; ++r) {
    float inv = 1.0f / lrun[r];
    int s = qBase + r + 8 * halfId;
    Out[((size_t)(nCol +  0) * S_LEN + s) * B_N + b] = o0[r] * inv;
    Out[((size_t)(nCol + 16) * S_LEN + s) * B_N + b] = o1[r] * inv;
    Out[((size_t)(nCol + 32) * S_LEN + s) * B_N + b] = o2[r] * inv;
    Out[((size_t)(nCol + 48) * S_LEN + s) * B_N + b] = o3[r] * inv;
  }
}

extern "C" void kernel_launch(void* const* d_in, const int* in_sizes, int n_in,
                              void* d_out, int out_size, void* d_ws, size_t ws_size,
                              hipStream_t stream) {
  const float* Q = (const float*)d_in[0];
  const float* K = (const float*)d_in[1];
  const float* V = (const float*)d_in[2];
  float* Out = (float*)d_out;
  (void)in_sizes; (void)n_in; (void)out_size;

  const float scale = 1.0f / sqrtf((float)S_LEN);  // d_k == S == 2048
  dim3 grid(B_N * (S_LEN / QT / WAVES));           // 512 blocks
  dim3 block(32 * WAVES);                          // 4 waves (wave32)

  if (ws_size >= (size_t)WS_BYTES) {
    _Float16* ws = (_Float16*)d_ws;
    // pre-pass: 3 * 2^20 fp16 pairs, 256 threads each
    cvt_kernel<<<dim3(3u * (1u << 20) / 256u), dim3(256), 0, stream>>>(Q, K, V, ws, scale);
    fa_ws_kernel<<<grid, block, 0, stream>>>(ws, Out);
  } else {
    fa_fallback_kernel<<<grid, block, 0, stream>>>(Q, K, V, Out, scale);
  }
}
